// WeightGeneratorWithAtt_5806795784246
// MI455X (gfx1250) — matrix operand order repaired and checked by reading, not compiled, not run on hardware
//
#include <hip/hip_runtime.h>
#include <math.h>

// ---------------- CDNA5 WMMA types ----------------
typedef __attribute__((ext_vector_type(16))) __bf16 v16bf;
typedef __attribute__((ext_vector_type(8)))  float  v8f;

#define BSZ 64
#define NN  2048
#define CC  256
#define NB  100
#define NBP 112          // 7 tiles of 16
#define EPSF 1e-12f

#define ROWS_PER_BLOCK 128   // 8 waves x 16 rows
#define BLOCKS_PER_B   16    // 2048 / 128

// fp32 -> bf16 via native convert (RNE); lowers to v_cvt_pk_bf16_f32 pairs
__device__ __forceinline__ __bf16 f2bf(float f) { return (__bf16)f; }

union ABOp { v16bf v; unsigned u[8]; };

__device__ __forceinline__ v8f v8f_splat(float x) {
    v8f z;
    #pragma unroll
    for (int i = 0; i < 8; ++i) z[i] = x;
    return z;
}

// load one 16x(32-slice) B tile (K = base+{0..7, 16..23} bf16 per lane) into an operand
__device__ __forceinline__ void load_bt(ABOp& Bv, const __bf16* base) {
    uint4 b0 = *(const uint4*)(base);
    uint4 b1 = *(const uint4*)(base + 16);
    Bv.u[0] = b0.x; Bv.u[1] = b0.y; Bv.u[2] = b0.z; Bv.u[3] = b0.w;
    Bv.u[4] = b1.x; Bv.u[5] = b1.y; Bv.u[6] = b1.z; Bv.u[7] = b1.w;
}

// ---------------- prep kernels ----------------

__global__ void __launch_bounds__(256)
zero_kernel(float* att_sum, float* sum_x) {
    int i = blockIdx.x * 256 + threadIdx.x;
    if (i < BSZ * NBP) att_sum[i] = 0.f;
    else {
        int j = i - BSZ * NBP;
        if (j < BSZ * CC) sum_x[j] = 0.f;
    }
}

__global__ void __launch_bounds__(256)
wq_cvt_kernel(const float* __restrict__ Wq, __bf16* __restrict__ wq_bf) {
    int i = blockIdx.x * 256 + threadIdx.x;   // grid covers 256*256
    wq_bf[i] = f2bf(Wq[i]);
}

// kn = l2norm(base_weights @ Wk^T + bk), rows padded to NBP with zeros, bf16
__global__ void __launch_bounds__(256)
kn_kernel(const float* __restrict__ bw, const float* __restrict__ Wk,
          const float* __restrict__ bk, __bf16* __restrict__ kn_bf) {
    __shared__ float bwrow[CC];
    __shared__ float red[CC];
    const int m = blockIdx.x, c = threadIdx.x;
    if (m >= NB) { kn_bf[m * CC + c] = (__bf16)0.f; return; }
    bwrow[c] = bw[m * CC + c];
    __syncthreads();
    float d = bk[c];
    const float* wkr = Wk + (size_t)c * CC;
    #pragma unroll 8
    for (int j = 0; j < CC; ++j) d = fmaf(bwrow[j], wkr[j], d);
    red[c] = d * d;
    __syncthreads();
    for (int s = 128; s > 0; s >>= 1) {
        if (c < s) red[c] += red[c + s];
        __syncthreads();
    }
    float sc = 1.f / fmaxf(sqrtf(red[0]), EPSF);
    kn_bf[m * CC + c] = f2bf(d * sc);
}

__global__ void __launch_bounds__(256)
denom_kernel(const float* __restrict__ mask, float* __restrict__ denom) {
    __shared__ float red[256];
    const int b = blockIdx.x, t = threadIdx.x;
    float s = 0.f;
    for (int i = t; i < NN; i += 256) s += mask[(size_t)b * NN + i];
    red[t] = s;
    __syncthreads();
    for (int st = 128; st > 0; st >>= 1) {
        if (t < st) red[t] += red[t + st];
        __syncthreads();
    }
    if (t == 0) denom[b] = fmaxf(red[0], EPSF);
}

// ---------------- main WMMA kernel ----------------
// block: 256 threads (8 waves), processes 128 rows of one batch b.
// wave w owns rows [w*16, w*16+16) of the block.

__global__ void __launch_bounds__(256)
wg_att_main(const float* __restrict__ X, const float* __restrict__ mask,
            const __bf16* __restrict__ wq_bf,
            const __bf16* __restrict__ kn_bf,
            const float* __restrict__ bq,
            float* __restrict__ att_sum, float* __restrict__ sum_x)
{
    extern __shared__ char smem[];
    __bf16* qn    = (__bf16*)smem;                                           // [128][256] bf16
    float* logits = (float*)(smem + ROWS_PER_BLOCK * CC * 2);                // [128][112] f32
    float* maskv  = (float*)(smem + ROWS_PER_BLOCK * CC * 2
                                  + ROWS_PER_BLOCK * NBP * 4);               // [128]

    const int tid  = threadIdx.x;
    const int b    = blockIdx.x >> 4;
    const int nblk = blockIdx.x & (BLOCKS_PER_B - 1);
    const long rowg0 = (long)b * NN + (long)nblk * ROWS_PER_BLOCK;

    const int wave = tid >> 5;
    const int lane = tid & 31;
    const int lrow = lane & 15;     // A: row within tile / B: column within tile
    const int lhi  = lane >> 4;     // half-wave select
    const int rowL = wave * 16 + lrow;
    const float* xrow = X + (rowg0 + rowL) * CC;

    // warm WGP$/L2 for this wave's GEMM1 A rows (global_prefetch_b8)
    __builtin_prefetch(xrow, 0, 3);
    __builtin_prefetch(xrow + 128, 0, 3);

    if (tid < ROWS_PER_BLOCK) maskv[tid] = mask[rowg0 + tid];
    __syncthreads();

    // ---- branch 1: masked column sums of X (coalesced; warms caches for GEMM1) ----
    {
        const float* xc = X + rowg0 * CC + tid;
        float s = 0.f;
        #pragma unroll 4
        for (int r = 0; r < ROWS_PER_BLOCK; ++r) s = fmaf(maskv[r], xc[(long)r * CC], s);
        atomicAdd(&sum_x[b * CC + tid], s);
    }

    // ---- GEMM1: q[16 x 256] = X_tile @ Wq^T + bq, bf16 WMMA, f32 acc ----
    // bias folded into the accumulator init: D = A*B + C with C = bq broadcast.
    v8f acc[16];
    #pragma unroll
    for (int nt = 0; nt < 16; ++nt) acc[nt] = v8f_splat(bq[nt * 16 + lrow]);

    #pragma unroll
    for (int kt = 0; kt < 8; ++kt) {
        // A operand: lane(<16): K = kt*32 + {0..7, 16..23}; lane(>=16): +8
        const int kb = kt * 32 + lhi * 8;
        float4 a0 = *(const float4*)(xrow + kb);
        float4 a1 = *(const float4*)(xrow + kb + 4);
        float4 a2 = *(const float4*)(xrow + kb + 16);
        float4 a3 = *(const float4*)(xrow + kb + 20);
        v16bf Av;
        Av[0]  = f2bf(a0.x); Av[1]  = f2bf(a0.y); Av[2]  = f2bf(a0.z); Av[3]  = f2bf(a0.w);
        Av[4]  = f2bf(a1.x); Av[5]  = f2bf(a1.y); Av[6]  = f2bf(a1.z); Av[7]  = f2bf(a1.w);
        Av[8]  = f2bf(a2.x); Av[9]  = f2bf(a2.y); Av[10] = f2bf(a2.z); Av[11] = f2bf(a2.w);
        Av[12] = f2bf(a3.x); Av[13] = f2bf(a3.y); Av[14] = f2bf(a3.z); Av[15] = f2bf(a3.w);

        // B operand: lane = output column n; K = lhi*8 + {0..7, 16..23}.
        // Software-pipelined 3-deep rotating buffer: tile nt+3 loads while nt computes.
        const __bf16* wqk = wq_bf + (size_t)lrow * CC + kt * 32 + lhi * 8;
        ABOp Bt[3];
        load_bt(Bt[0], wqk + (size_t)0 * 16 * CC);
        load_bt(Bt[1], wqk + (size_t)1 * 16 * CC);
        load_bt(Bt[2], wqk + (size_t)2 * 16 * CC);
        #pragma unroll
        for (int nt = 0; nt < 16; ++nt) {
            acc[nt] = __builtin_amdgcn_wmma_f32_16x16x32_bf16(
                false, Av, false, Bt[nt % 3].v, (short)0, acc[nt], false, false);
            if (nt + 3 < 16)
                load_bt(Bt[nt % 3], wqk + (size_t)(nt + 3) * 16 * CC);
        }
    }

    // ---- row norms (C layout: lane = column, VGPR r = row r + 8*lhi) ----
    float n2[8];
    #pragma unroll
    for (int r = 0; r < 8; ++r) n2[r] = 0.f;
    #pragma unroll
    for (int nt = 0; nt < 16; ++nt)
        #pragma unroll
        for (int r = 0; r < 8; ++r) n2[r] = fmaf(acc[nt][r], acc[nt][r], n2[r]);
    #pragma unroll
    for (int m = 1; m < 16; m <<= 1)
        #pragma unroll
        for (int r = 0; r < 8; ++r) n2[r] += __shfl_xor(n2[r], m, 32);
    float sc[8];
    #pragma unroll
    for (int r = 0; r < 8; ++r) sc[r] = 1.f / fmaxf(sqrtf(n2[r]), EPSF);

    // ---- stage normalized q as bf16 into LDS (transpose C-layout -> A-layout) ----
    #pragma unroll
    for (int nt = 0; nt < 16; ++nt)
        #pragma unroll
        for (int r = 0; r < 8; ++r)
            qn[(wave * 16 + r + 8 * lhi) * CC + nt * 16 + lrow] = f2bf(acc[nt][r] * sc[r]);
    // same-wave LDS ops are in-order: safe to read our own rows below.

    // ---- GEMM2: logits[16 x 112] = qn @ kn^T ----
    v8f acc2[7];
    #pragma unroll
    for (int mt = 0; mt < 7; ++mt) acc2[mt] = v8f_splat(0.f);

    #pragma unroll
    for (int kt = 0; kt < 8; ++kt) {
        const __bf16* qr = qn + (size_t)(wave * 16 + lrow) * CC + kt * 32 + lhi * 8;
        uint4 a0 = *(const uint4*)(qr);
        uint4 a1 = *(const uint4*)(qr + 16);
        ABOp A;
        A.u[0] = a0.x; A.u[1] = a0.y; A.u[2] = a0.z; A.u[3] = a0.w;
        A.u[4] = a1.x; A.u[5] = a1.y; A.u[6] = a1.z; A.u[7] = a1.w;

        // 2-deep pipelined kn tiles
        const __bf16* knk = kn_bf + (size_t)lrow * CC + kt * 32 + lhi * 8;
        ABOp Bt[2];
        load_bt(Bt[0], knk + (size_t)0 * 16 * CC);
        load_bt(Bt[1], knk + (size_t)1 * 16 * CC);
        #pragma unroll
        for (int mt = 0; mt < 7; ++mt) {
            acc2[mt] = __builtin_amdgcn_wmma_f32_16x16x32_bf16(
                false, A.v, false, Bt[mt & 1].v, (short)0, acc2[mt], false, false);
            if (mt + 2 < 7)
                load_bt(Bt[mt & 1], knk + (size_t)(mt + 2) * 16 * CC);
        }
    }

    // ---- logits to LDS ----
    #pragma unroll
    for (int mt = 0; mt < 7; ++mt)
        #pragma unroll
        for (int r = 0; r < 8; ++r)
            logits[(wave * 16 + r + 8 * lhi) * NBP + mt * 16 + lrow] = acc2[mt][r];
    __syncthreads();

    // ---- per-row softmax * mask, written in place (cols [100,112) untouched) ----
    if (tid < ROWS_PER_BLOCK) {
        float* row = logits + tid * NBP;
        float mx = -1e30f;
        #pragma unroll 4
        for (int m = 0; m < NB; ++m) mx = fmaxf(mx, row[m]);
        float s = 0.f;
        #pragma unroll 4
        for (int m = 0; m < NB; ++m) s += __expf(row[m] - mx);
        float inv = maskv[tid] / s;
        #pragma unroll 4
        for (int m = 0; m < NB; ++m) row[m] = __expf(row[m] - mx) * inv;
    }
    __syncthreads();

    // ---- column-reduce masked attention, accumulate into att_sum[b] ----
    if (tid < NB) {
        float s = 0.f;
        #pragma unroll 4
        for (int r = 0; r < ROWS_PER_BLOCK; ++r) s += logits[r * NBP + tid];
        atomicAdd(&att_sum[b * NBP + tid], s);
    }
}

// ---------------- final combine ----------------
__global__ void __launch_bounds__(256)
final_kernel(const float* __restrict__ bw, const float* __restrict__ att_sum,
             const float* __restrict__ sum_x, const float* __restrict__ denom,
             const float* __restrict__ w_avg, const float* __restrict__ w_att,
             float* __restrict__ out)
{
    __shared__ float att[NB];
    const int b = blockIdx.x, c = threadIdx.x;
    if (c < NB) att[c] = att_sum[b * NBP + c];
    __syncthreads();
    const float dn = denom[b];
    float wn2 = 0.f;
    #pragma unroll 4
    for (int m = 0; m < NB; ++m) wn2 = fmaf(att[m], bw[m * CC + c], wn2);
    float wn1 = (sum_x[b * CC + c] / dn) * w_avg[c];
    out[b * CC + c] = 0.5f * (wn1 + (wn2 / dn) * w_att[c]);
}

// ---------------- launch ----------------
extern "C" void kernel_launch(void* const* d_in, const int* in_sizes, int n_in,
                              void* d_out, int out_size, void* d_ws, size_t ws_size,
                              hipStream_t stream)
{
    const float* base_w = (const float*)d_in[0];
    const float* X      = (const float*)d_in[1];
    const float* mask   = (const float*)d_in[2];
    const float* Wq     = (const float*)d_in[3];
    const float* bq     = (const float*)d_in[4];
    const float* Wk     = (const float*)d_in[5];
    const float* bk     = (const float*)d_in[6];
    const float* w_avg  = (const float*)d_in[7];
    const float* w_att  = (const float*)d_in[8];
    float* out = (float*)d_out;

    char* ws = (char*)d_ws;
    __bf16* wq_bf = (__bf16*)(ws);                                      // 256*256*2 = 131072
    __bf16* kn_bf = (__bf16*)(ws + 131072);                             // 112*256*2 =  57344
    float* att_sum = (float*)(ws + 131072 + 57344);                     // 64*112*4  =  28672
    float* sum_x   = (float*)(ws + 131072 + 57344 + 28672);             // 64*256*4  =  65536
    float* denom   = (float*)(ws + 131072 + 57344 + 28672 + 65536);     // 64*4

    zero_kernel<<<(BSZ * NBP + BSZ * CC + 255) / 256, 256, 0, stream>>>(att_sum, sum_x);
    wq_cvt_kernel<<<(CC * CC) / 256, 256, 0, stream>>>(Wq, wq_bf);
    kn_kernel<<<NBP, 256, 0, stream>>>(base_w, Wk, bk, kn_bf);
    denom_kernel<<<BSZ, 256, 0, stream>>>(mask, denom);

    const size_t smem = ROWS_PER_BLOCK * CC * 2      // qn bf16
                      + ROWS_PER_BLOCK * NBP * 4     // logits f32
                      + ROWS_PER_BLOCK * 4;          // mask cache
    wg_att_main<<<BSZ * BLOCKS_PER_B, 256, smem, stream>>>(
        X, mask, wq_bf, kn_bf, bq, att_sum, sum_x);

    final_kernel<<<BSZ, 256, 0, stream>>>(base_w, att_sum, sum_x, denom, w_avg, w_att, out);
}
